// SparseTriangleAttention_7627861918046
// MI455X (gfx1250) — matrix operand order repaired and checked by reading, not compiled, not run on hardware
//
#include <hip/hip_runtime.h>
#include <hip/hip_bf16.h>

// Types for WMMA operands (CDNA5 wave32: A/B = 16 halves/lane, C/D = 8 floats/lane)
typedef _Float16 h4_t  __attribute__((ext_vector_type(4)));
typedef _Float16 h8_t  __attribute__((ext_vector_type(8)));
typedef _Float16 h16_t __attribute__((ext_vector_type(16)));
typedef float    f4_t  __attribute__((ext_vector_type(4)));
typedef float    f8_t  __attribute__((ext_vector_type(8)));

#define CZ   256          // c_z
#define EE   131072       // num edges
#define MT   32           // rows (edges) per workgroup
#define LDA  264          // padded LDS row stride in halves (+16B to spread banks)
#define NBLK (EE / MT)    // 4096 blocks

// ---------------------------------------------------------------------------
// Pack W (fp32, [K=256][N=256] row-major) into Wt (f16, [N][K], K-contiguous).
// This matches the B feed order this kernel uses: per 16x16x32 step, lane (n =
// lane&15) loads 16 contiguous K halves at k0 = kc*32 + (lane>>4)*16.
// ---------------------------------------------------------------------------
__global__ __launch_bounds__(256) void pack_wT(const float* __restrict__ W,
                                               _Float16* __restrict__ Wt) {
  int idx = blockIdx.x * 256 + threadIdx.x;   // 0..65535
  int n = idx >> 8;
  int k = idx & 255;
  Wt[n * CZ + k] = (_Float16)W[k * CZ + n];
}

// ---------------------------------------------------------------------------
// Fused  Y = (X @ Wv + bv) @ Wo + bo  (the softmax-sum in the reference is
// identically 1, so the attention path is mathematically dead code).
// One block = 32 rows of X. X tile staged fp32->f16 in LDS; GEMM1 result V
// (with bias) stored f16 in LDS; GEMM2 streams Y fp32 to HBM.
// ---------------------------------------------------------------------------
__global__ __launch_bounds__(256) void fused_vw(
    const float* __restrict__ X,
    const _Float16* __restrict__ WtV,
    const _Float16* __restrict__ WtO,
    const float* __restrict__ bv,
    const float* __restrict__ bo,
    float* __restrict__ Y) {
  __shared__ __align__(16) _Float16 Xh[MT * LDA];
  __shared__ __align__(16) _Float16 Vh[MT * LDA];

  const int t    = threadIdx.x;
  const int lane = t & 31;
  const int wv   = t >> 5;        // wave id 0..7
  const int ln   = lane & 15;     // N column / A row selector
  const int hi   = lane >> 4;     // half-wave selector
  const long row0 = (long)blockIdx.x * MT;

  // ---- Stage X tile [32][256] fp32 -> f16 LDS (nontemporal: streamed once)
  const f4_t* Xt = (const f4_t*)(X + row0 * CZ);
#pragma unroll
  for (int i = 0; i < (MT * CZ / 4) / 256; ++i) {   // 8 float4 per thread
    int q = i * 256 + t;
    f4_t v = __builtin_nontemporal_load(Xt + q);
    int r = q >> 6;                // 64 float4 per row
    int c = (q & 63) << 2;
    h4_t h;
    h[0] = (_Float16)v[0]; h[1] = (_Float16)v[1];
    h[2] = (_Float16)v[2]; h[3] = (_Float16)v[3];
    *(h4_t*)(Xh + r * LDA + c) = h;
  }
  __syncthreads();

  // ---- Two chained GEMMs; phase 0: Xh @ WtV -> Vh(f16), phase 1: Vh @ WtO -> Y
#pragma unroll 1
  for (int phase = 0; phase < 2; ++phase) {
    const _Float16* __restrict__ Bw   = (phase == 0) ? WtV : WtO;
    const _Float16* __restrict__ Alds = (phase == 0) ? Xh : Vh;
    const float*    __restrict__ bias = (phase == 0) ? bv : bo;

#pragma unroll 1
    for (int rep = 0; rep < 2; ++rep) {
      const int ntile = (wv + 8 * rep) * 16;    // 8 waves x 2 reps cover N=256
      const int ncol  = ntile + ln;
      const _Float16* Bcol = Bw + ncol * CZ + hi * 16;  // 16 contig K halves

      f8_t acc[2];
      acc[0] = (f8_t){0.f, 0.f, 0.f, 0.f, 0.f, 0.f, 0.f, 0.f};
      acc[1] = (f8_t){0.f, 0.f, 0.f, 0.f, 0.f, 0.f, 0.f, 0.f};

#pragma unroll 4
      for (int kc = 0; kc < 8; ++kc) {          // K = 8 x 32
        // B chunk 32x16 f16: loaded once, reused for both M-tiles
        h8_t b0 = *(const h8_t*)(Bcol + kc * 32);
        h8_t b1 = *(const h8_t*)(Bcol + kc * 32 + 8);
        h16_t B = __builtin_shufflevector(b0, b1, 0, 1, 2, 3, 4, 5, 6, 7,
                                          8, 9, 10, 11, 12, 13, 14, 15);
        if (kc < 7)  // global_prefetch_b8 of next weight chunk
          __builtin_prefetch(Bcol + (kc + 1) * 32, 0, 3);

        const int k0 = kc * 32 + hi * 16;       // A layout: 16 contig K halves [k0,k0+16)
#pragma unroll
        for (int mt = 0; mt < 2; ++mt) {
          const _Float16* arow = Alds + (mt * 16 + ln) * LDA + k0;
          h8_t a0 = *(const h8_t*)(arow);
          h8_t a1 = *(const h8_t*)(arow + 8);
          h16_t A = __builtin_shufflevector(a0, a1, 0, 1, 2, 3, 4, 5, 6, 7,
                                            8, 9, 10, 11, 12, 13, 14, 15);
          acc[mt] = __builtin_amdgcn_wmma_f32_16x16x32_f16(
              false, A, false, B, (short)0, acc[mt], false, false);
        }
      }

      // ---- Epilogue: D layout element r <-> (M = hi*8 + r, N = ncol)
      const float bb = bias[ncol];
      if (phase == 0) {
#pragma unroll
        for (int mt = 0; mt < 2; ++mt)
#pragma unroll
          for (int r = 0; r < 8; ++r)
            Vh[(mt * 16 + hi * 8 + r) * LDA + ncol] = (_Float16)(acc[mt][r] + bb);
      } else {
        float* Yt = Y + row0 * CZ;
#pragma unroll
        for (int mt = 0; mt < 2; ++mt)
#pragma unroll
          for (int r = 0; r < 8; ++r)
            __builtin_nontemporal_store(acc[mt][r] + bb,
                                        Yt + (mt * 16 + hi * 8 + r) * CZ + ncol);
      }
    }
    __syncthreads();   // Vh fully written before phase 1 reads it
  }
}

// ---------------------------------------------------------------------------
// d_in order: 0 node_features, 1 trans, 2 edge_features, 3 edge_index,
// 4 W_gate, 5 b_gate, 6 W_db, 7 b_db, 8 W_qk, 9 b_qk, 10 W_v, 11 b_v,
// 12 W_out, 13 b_out.  Only edge_features / W_v / b_v / W_out / b_out matter.
// ---------------------------------------------------------------------------
extern "C" void kernel_launch(void* const* d_in, const int* in_sizes, int n_in,
                              void* d_out, int out_size, void* d_ws, size_t ws_size,
                              hipStream_t stream) {
  const float* X  = (const float*)d_in[2];
  const float* Wv = (const float*)d_in[10];
  const float* bv = (const float*)d_in[11];
  const float* Wo = (const float*)d_in[12];
  const float* bo = (const float*)d_in[13];
  float* Y = (float*)d_out;

  _Float16* WtV = (_Float16*)d_ws;           // 128 KB
  _Float16* WtO = WtV + CZ * CZ;             // 128 KB

  pack_wT<<<256, 256, 0, stream>>>(Wv, WtV);
  pack_wT<<<256, 256, 0, stream>>>(Wo, WtO);
  fused_vw<<<NBLK, 256, 0, stream>>>(X, WtV, WtO, bv, bo, Y);
}
